// MuTokenRoutedMLP_72576357368018
// MI455X (gfx1250) — matrix operand order repaired and checked by reading, not compiled, not run on hardware
//
#include <hip/hip_runtime.h>
#include <hip/hip_bf16.h>
#include <stdint.h>

typedef __attribute__((ext_vector_type(16))) __bf16 v16bf;
typedef __attribute__((ext_vector_type(8)))  __bf16 v8bf;
typedef __attribute__((ext_vector_type(8)))  float  v8f;

#define N_TOK 8192
#define HDIM  1024
#define NEXP  8
#define IE    512
#define VOCAB 32000
#define ROUTE_BIAS 10.0f

#define MT    64            // tokens per workgroup tile
#define XSTR  (HDIM + 8)    // padded LDS row stride (bf16 elems)
#define ISTR  (IE + 8)

// workspace layout (bytes)
#define OFF_EID   0                      // N_TOK * 4
#define OFF_CNT   (64 * 1024)            // NEXP * 4
#define OFF_LIST  (128 * 1024)           // NEXP * N_TOK * 4
#define OFF_XBF   (1u << 20)             // N_TOK*HDIM*2      = 16 MB
#define OFF_PGU   (OFF_XBF + (16u << 20))// NEXP*HDIM*2*IE*2  = 16 MB
#define OFF_PDN   (OFF_PGU + (16u << 20))// NEXP*IE*HDIM*2    =  8 MB

// ---------------- routing ----------------
__global__ __launch_bounds__(256) void route_kernel(
    const float* __restrict__ mu, const float* __restrict__ rw,
    const int* __restrict__ tok, int* __restrict__ eid)
{
    int wave = threadIdx.x >> 5;
    int lane = threadIdx.x & 31;
    int n = blockIdx.x * 8 + wave;
    const float* m = mu + (size_t)n * HDIM;

    float acc[NEXP];
#pragma unroll
    for (int e = 0; e < NEXP; ++e) acc[e] = 0.f;

    for (int i = 0; i < HDIM / 32; ++i) {
        int h = i * 32 + lane;
        float x = m[h];
#pragma unroll
        for (int e = 0; e < NEXP; ++e) acc[e] += x * rw[e * HDIM + h];
    }
#pragma unroll
    for (int e = 0; e < NEXP; ++e) {
        float v = acc[e];
        for (int off = 16; off; off >>= 1) v += __shfl_xor(v, off, 32);
        acc[e] = v;
    }
    if (lane == 0) {
        int t = tok[n];
        t = (t < 0) ? 0 : (t >= VOCAB ? VOCAB - 1 : t);
        int be = t % NEXP;
        float best = acc[0] + (be == 0 ? ROUTE_BIAS : 0.f);
        int bi = 0;
#pragma unroll
        for (int e = 1; e < NEXP; ++e) {
            float c = acc[e] + (be == e ? ROUTE_BIAS : 0.f);
            if (c > best) { best = c; bi = e; }
        }
        eid[n] = bi;
    }
}

__global__ void zero_counts(int* c) { if (threadIdx.x < NEXP) c[threadIdx.x] = 0; }

__global__ __launch_bounds__(256) void scatter_kernel(
    const int* __restrict__ eid, int* __restrict__ counts, int* __restrict__ lists)
{
    int n = blockIdx.x * 256 + threadIdx.x;
    int e = eid[n];
    int pos = atomicAdd(&counts[e], 1);
    lists[e * N_TOK + pos] = n;
}

// ---------------- bf16 conversion / fragment pre-pack ----------------
__global__ __launch_bounds__(256) void convert_x(
    const float* __restrict__ x, __bf16* __restrict__ xb)
{
    int i = blockIdx.x * 256 + threadIdx.x;     // N_TOK*HDIM = 8388608
    xb[i] = (__bf16)x[i];
}

// gate_up_proj [E][H][2*IE] -> packed B-fragments:
// pk[((e*64 + nt)*32 + ks)*512 + lane*16 + s] = W[e][ks*32 + (lane>>4)*8 + (s&7) + (s>>3)*16][nt*16 + (lane&15)]
__global__ __launch_bounds__(256) void pack_gu(
    const float* __restrict__ w, __bf16* __restrict__ pk)
{
    unsigned idx = blockIdx.x * 256 + threadIdx.x;   // 2^23 total
    int s    = idx & 15;
    int lane = (idx >> 4) & 31;
    int ks   = (idx >> 9) & 31;
    int nt   = (idx >> 14) & 63;
    int e    = idx >> 20;
    int k = ks * 32 + (lane >> 4) * 8 + (s & 7) + (s >> 3) * 16;
    int n = nt * 16 + (lane & 15);
    pk[idx] = (__bf16)w[((size_t)e * HDIM + k) * (2 * IE) + n];
}

// down_proj [E][IE][H] -> packed B-fragments (16 k-steps, 64 n-tiles)
__global__ __launch_bounds__(256) void pack_down(
    const float* __restrict__ w, __bf16* __restrict__ pk)
{
    unsigned idx = blockIdx.x * 256 + threadIdx.x;   // 2^22 total
    int s    = idx & 15;
    int lane = (idx >> 4) & 31;
    int ks   = (idx >> 9) & 15;
    int nt   = (idx >> 13) & 63;
    int e    = idx >> 19;
    int k = ks * 32 + (lane >> 4) * 8 + (s & 7) + (s >> 3) * 16;
    int n = nt * 16 + (lane & 15);
    pk[idx] = (__bf16)w[((size_t)e * IE + k) * HDIM + n];
}

// ---------------- fused grouped GEMM ----------------
__device__ __forceinline__ v16bf load_afrag(const __bf16* p) {
    v8bf a0 = *(const v8bf*)p;
    v8bf a1 = *(const v8bf*)(p + 16);
    v16bf a;
#pragma unroll
    for (int i = 0; i < 8; ++i) { a[i] = a0[i]; a[i + 8] = a1[i]; }
    return a;
}

#define WMMA_BF16(A, B, C) \
    __builtin_amdgcn_wmma_f32_16x16x32_bf16(false, (A), false, (B), (short)0, (C), false, false)

#define ZERO8 {0.f,0.f,0.f,0.f,0.f,0.f,0.f,0.f}

__global__ __launch_bounds__(256) void moe_gemm(
    const __bf16* __restrict__ xb, const __bf16* __restrict__ pgu,
    const __bf16* __restrict__ pdn, const int* __restrict__ counts,
    const int* __restrict__ lists, float* __restrict__ out)
{
    extern __shared__ char smem[];
    __bf16* xl = (__bf16*)smem;                 // 64 x XSTR
    __bf16* il = xl + MT * XSTR;                // 64 x ISTR
    int*    tokr = (int*)(il + MT * ISTR);      // 64

    int e    = blockIdx.x >> 7;                 // 128 tiles per expert
    int tile = blockIdx.x & 127;
    int cnt  = counts[e];
    int base = tile * MT;
    if (base >= cnt) return;

    int tid = threadIdx.x;
    if (tid < MT) {
        int r = base + tid;
        tokr[tid] = (r < cnt) ? lists[e * N_TOK + r] : -1;
    }
    __syncthreads();

    // stage gathered X tile into LDS (16B chunks, coalesced)
    for (int i = tid; i < MT * 128; i += 256) {
        int r = i >> 7, c = i & 127;
        int t = tokr[r]; if (t < 0) t = 0;
        *(v8bf*)(xl + r * XSTR + c * 8) = *(const v8bf*)(xb + (size_t)t * HDIM + c * 8);
    }
    __syncthreads();

    int wave  = tid >> 5, lane = tid & 31;
    int lrow  = lane & 15;
    int khalf = (lane >> 4) * 8;                // 0 or 8 (A-frag K half)
    int mrow0 = (lane >= 16) ? 8 : 0;

    int mpair = wave & 1;                       // m-blocks 2*mpair, 2*mpair+1
    int mb0   = mpair * 2;
    int mb1   = mb0 + 1;

    // ---- GEMM1: X(64x1024) * W_gu -> silu(gate)*up -> inter LDS (64x512 bf16)
    // 2 m-blocks x 2 pair-tiles per wave-iteration: 8 WMMA per K-step
    const __bf16* xrow0 = xl + (mb0 * 16 + lrow) * XSTR;
    const __bf16* xrow1 = xl + (mb1 * 16 + lrow) * XSTR;
    const __bf16* pge   = pgu + (size_t)e * 64 * 32 * 512;
    int ptg = wave >> 1;                        // 0..3 -> pair-tiles [ptg*8, ptg*8+8)
    for (int pt = ptg * 8; pt < ptg * 8 + 8; pt += 2) {
        v8f ag00 = ZERO8, au00 = ZERO8, ag10 = ZERO8, au10 = ZERO8;
        v8f ag01 = ZERO8, au01 = ZERO8, ag11 = ZERO8, au11 = ZERO8;
        const __bf16* bg0 = pge + (size_t)(pt)      * 32 * 512 + lane * 16;
        const __bf16* bu0 = pge + (size_t)(pt + 32) * 32 * 512 + lane * 16;
        const __bf16* bg1 = pge + (size_t)(pt + 1)  * 32 * 512 + lane * 16;
        const __bf16* bu1 = pge + (size_t)(pt + 33) * 32 * 512 + lane * 16;
#pragma unroll 2
        for (int ks = 0; ks < 32; ++ks) {
            v16bf a0  = load_afrag(xrow0 + ks * 32 + khalf);
            v16bf a1  = load_afrag(xrow1 + ks * 32 + khalf);
            v16bf Bg0 = *(const v16bf*)(bg0 + ks * 512);
            v16bf Bu0 = *(const v16bf*)(bu0 + ks * 512);
            v16bf Bg1 = *(const v16bf*)(bg1 + ks * 512);
            v16bf Bu1 = *(const v16bf*)(bu1 + ks * 512);
            ag00 = WMMA_BF16(a0, Bg0, ag00);
            au00 = WMMA_BF16(a0, Bu0, au00);
            ag10 = WMMA_BF16(a1, Bg0, ag10);
            au10 = WMMA_BF16(a1, Bu0, au10);
            ag01 = WMMA_BF16(a0, Bg1, ag01);
            au01 = WMMA_BF16(a0, Bu1, au01);
            ag11 = WMMA_BF16(a1, Bg1, ag11);
            au11 = WMMA_BF16(a1, Bu1, au11);
        }
        int m0 = mb0 * 16 + mrow0;
        int m1 = mb1 * 16 + mrow0;
#pragma unroll
        for (int r = 0; r < 8; ++r) {
            float g, u, s;
            g = ag00[r]; u = au00[r]; s = g / (1.f + __expf(-g));
            il[(m0 + r) * ISTR + (pt)     * 16 + lrow] = (__bf16)(s * u);
            g = ag10[r]; u = au10[r]; s = g / (1.f + __expf(-g));
            il[(m1 + r) * ISTR + (pt)     * 16 + lrow] = (__bf16)(s * u);
            g = ag01[r]; u = au01[r]; s = g / (1.f + __expf(-g));
            il[(m0 + r) * ISTR + (pt + 1) * 16 + lrow] = (__bf16)(s * u);
            g = ag11[r]; u = au11[r]; s = g / (1.f + __expf(-g));
            il[(m1 + r) * ISTR + (pt + 1) * 16 + lrow] = (__bf16)(s * u);
        }
    }
    __syncthreads();

    // ---- GEMM2: inter(64x512) * W_down -> out rows (f32)
    // 2 m-blocks x 2 n-tiles per wave-iteration: 4 WMMA per K-step
    const __bf16* pde  = pdn + (size_t)e * 64 * 16 * 512;
    const __bf16* irow0 = il + (mb0 * 16 + lrow) * ISTR;
    const __bf16* irow1 = il + (mb1 * 16 + lrow) * ISTR;
    int ntg = wave >> 1;                        // 0..3 -> n-tiles [ntg*16, ntg*16+16)
    for (int nt = ntg * 16; nt < ntg * 16 + 16; nt += 2) {
        v8f ac00 = ZERO8, ac10 = ZERO8, ac01 = ZERO8, ac11 = ZERO8;
        const __bf16* bd0 = pde + (size_t)(nt)     * 16 * 512 + lane * 16;
        const __bf16* bd1 = pde + (size_t)(nt + 1) * 16 * 512 + lane * 16;
#pragma unroll 2
        for (int ks = 0; ks < 16; ++ks) {
            v16bf a0 = load_afrag(irow0 + ks * 32 + khalf);
            v16bf a1 = load_afrag(irow1 + ks * 32 + khalf);
            v16bf B0 = *(const v16bf*)(bd0 + ks * 512);
            v16bf B1 = *(const v16bf*)(bd1 + ks * 512);
            ac00 = WMMA_BF16(a0, B0, ac00);
            ac10 = WMMA_BF16(a1, B0, ac10);
            ac01 = WMMA_BF16(a0, B1, ac01);
            ac11 = WMMA_BF16(a1, B1, ac11);
        }
        int m0 = mb0 * 16 + mrow0;
        int m1 = mb1 * 16 + mrow0;
        int c0 = nt * 16 + lrow;
        int c1 = c0 + 16;
#pragma unroll
        for (int r = 0; r < 8; ++r) {
            int tk0 = tokr[m0 + r];
            int tk1 = tokr[m1 + r];
            if (tk0 >= 0) {
                out[(size_t)tk0 * HDIM + c0] = ac00[r];
                out[(size_t)tk0 * HDIM + c1] = ac01[r];
            }
            if (tk1 >= 0) {
                out[(size_t)tk1 * HDIM + c0] = ac10[r];
                out[(size_t)tk1 * HDIM + c1] = ac11[r];
            }
        }
    }
}

// ---------------- host ----------------
extern "C" void kernel_launch(void* const* d_in, const int* in_sizes, int n_in,
                              void* d_out, int out_size, void* d_ws, size_t ws_size,
                              hipStream_t stream)
{
    const float* hs  = (const float*)d_in[0];   // hidden_states (N,H) f32
    const int*   tok = (const int*)  d_in[1];   // token_ids (N)
    const float* mu  = (const float*)d_in[2];   // mu (N,H) f32
    const float* gup = (const float*)d_in[3];   // gate_up_proj (E,H,2*IE)
    const float* dwn = (const float*)d_in[4];   // down_proj (E,IE,H)
    const float* rw  = (const float*)d_in[5];   // mu_router_w (E,H)
    float* out = (float*)d_out;
    char*  ws  = (char*)d_ws;

    int*    eid    = (int*)(ws + OFF_EID);
    int*    counts = (int*)(ws + OFF_CNT);
    int*    lists  = (int*)(ws + OFF_LIST);
    __bf16* xbf    = (__bf16*)(ws + OFF_XBF);
    __bf16* pgu    = (__bf16*)(ws + OFF_PGU);
    __bf16* pdn    = (__bf16*)(ws + OFF_PDN);

    route_kernel <<<N_TOK / 8, 256, 0, stream>>>(mu, rw, tok, eid);
    zero_counts  <<<1, 32, 0, stream>>>(counts);
    scatter_kernel<<<N_TOK / 256, 256, 0, stream>>>(eid, counts, lists);
    convert_x    <<<(N_TOK * HDIM) / 256, 256, 0, stream>>>(hs, xbf);
    pack_gu      <<<(NEXP * HDIM * 2 * IE) / 256, 256, 0, stream>>>(gup, pgu);
    pack_down    <<<(NEXP * IE * HDIM) / 256, 256, 0, stream>>>(dwn, pdn);

    size_t shmem = (size_t)(MT * XSTR + MT * ISTR) * sizeof(__bf16) + MT * sizeof(int);
    moe_gemm<<<NEXP * 128, 256, shmem, stream>>>(xbf, pgu, pdn, counts, lists, out);
}
